// AnomalyAttention_34875134443770
// MI455X (gfx1250) — matrix operand order repaired and checked by reading, not compiled, not run on hardware
//
#include <hip/hip_runtime.h>

typedef __bf16 bf16_t;
typedef __attribute__((ext_vector_type(16))) __bf16 v16bf;
typedef __attribute__((ext_vector_type(8)))  __bf16 v8bf;
typedef __attribute__((ext_vector_type(8)))  float   v8f;

constexpr int kB = 4;
constexpr int kN = 2048;
constexpr int kD = 512;                       // D == D_MODEL
constexpr size_t XN = (size_t)kB * kN * kD;   // elements in x / Q / K / V
constexpr size_t WN = (size_t)kD * kD;        // elements in each weight matrix

// attention LDS tiling (elements)
constexpr int KT_ELEM = 32 * kD;              // 32 keys x 512 feat  (32 KB)
constexpr int VT_ELEM = 256 * 32;             // 256 cols x 32 keys  (16 KB)
constexpr int SP_ELEM = 4 * 16 * 32;          // per-wave S staging   (4 KB)
constexpr size_t SMEM_BYTES = (2 * KT_ELEM + 2 * VT_ELEM + SP_ELEM) * sizeof(unsigned short); // 102400

__device__ __forceinline__ v16bf cat8(v8bf lo, v8bf hi) {
  v16bf r;
#pragma unroll
  for (int i = 0; i < 8; ++i) { r[i] = lo[i]; r[i + 8] = hi[i]; }
  return r;
}

// ---------------- conversion kernels ----------------
__global__ void cvt_x_kernel(const float* __restrict__ x, unsigned short* __restrict__ xb_) {
  bf16_t* xb = (bf16_t*)xb_;
  size_t i = (size_t)blockIdx.x * blockDim.x + threadIdx.x;
  xb[i] = (bf16_t)x[i];
}

// Writes W^T (shape d_model x D) in bf16 for Wq/Wk/Wv so WMMA B-fragments are contiguous.
__global__ void cvt_w_kernel(const float* __restrict__ Wq, const float* __restrict__ Wk,
                             const float* __restrict__ Wv,
                             unsigned short* __restrict__ wqt_, unsigned short* __restrict__ wkt_,
                             unsigned short* __restrict__ wvt_) {
  bf16_t* wqt = (bf16_t*)wqt_;
  bf16_t* wkt = (bf16_t*)wkt_;
  bf16_t* wvt = (bf16_t*)wvt_;
  size_t i = (size_t)blockIdx.x * blockDim.x + threadIdx.x;  // out index: c*kD + k
  int c = (int)(i >> 9);          // kD == 512
  int k = (int)(i & (kD - 1));
  size_t src = (size_t)k * kD + c;
  wqt[i] = (bf16_t)Wq[src];
  wkt[i] = (bf16_t)Wk[src];
  wvt[i] = (bf16_t)Wv[src];
}

// ---------------- sigma = x @ Ws ----------------
__global__ void sigma_kernel(const float* __restrict__ x, const float* __restrict__ Ws,
                             float* __restrict__ sig) {
  int row  = blockIdx.x * (blockDim.x >> 5) + (threadIdx.x >> 5);
  int lane = threadIdx.x & 31;
  const float* xr = x + (size_t)row * kD;
  float s = 0.f;
  for (int k = lane; k < kD; k += 32) s += xr[k] * Ws[k];
#pragma unroll
  for (int m = 16; m >= 1; m >>= 1) s += __shfl_xor(s, m, 32);
  if (lane == 0) sig[row] = s;
}

// ---------------- Q/K/V projections via WMMA bf16 ----------------
__global__ __launch_bounds__(128) void proj_kernel(
    const unsigned short* __restrict__ xb_,  const unsigned short* __restrict__ wqt_,
    const unsigned short* __restrict__ wkt_, const unsigned short* __restrict__ wvt_,
    unsigned short* __restrict__ qb_, unsigned short* __restrict__ kb_,
    unsigned short* __restrict__ vt_) {
  const bf16_t* xb  = (const bf16_t*)xb_;
  const bf16_t* wqt = (const bf16_t*)wqt_;
  const bf16_t* wkt = (const bf16_t*)wkt_;
  const bf16_t* wvt = (const bf16_t*)wvt_;
  bf16_t* qb = (bf16_t*)qb_;
  bf16_t* kb = (bf16_t*)kb_;
  bf16_t* vt = (bf16_t*)vt_;

  const int wave = (blockIdx.x * blockDim.x + threadIdx.x) >> 5;
  const int lane = threadIdx.x & 31;
  const int ntc  = kD / 16;            // 32 col tiles
  const int row0 = (wave / ntc) * 16;  // over B*N rows
  const int col0 = (wave % ntc) * 16;
  const int g = lane >> 4, ln = lane & 15;

  v8f cq = {}, ck = {}, cv = {};
  const bf16_t* arow = xb + (size_t)(row0 + ln) * kD;  // A: row M = lane%16
  const size_t  bro  = (size_t)(col0 + ln) * kD;       // B: col N = lane%16 (from W^T)
  for (int k0 = 0; k0 < kD; k0 += 32) {
    v8bf alo = *(const v8bf*)(arow + k0 + 16 * g);
    v8bf ahi = *(const v8bf*)(arow + k0 + 16 * g + 8);
    v16bf a = cat8(alo, ahi);
    v16bf bq = *(const v16bf*)(wqt + bro + k0 + 16 * g);
    v16bf bk = *(const v16bf*)(wkt + bro + k0 + 16 * g);
    v16bf bv = *(const v16bf*)(wvt + bro + k0 + 16 * g);
    cq = __builtin_amdgcn_wmma_f32_16x16x32_bf16(false, a, false, bq, (short)0, cq, false, false);
    ck = __builtin_amdgcn_wmma_f32_16x16x32_bf16(false, a, false, bk, (short)0, ck, false, false);
    cv = __builtin_amdgcn_wmma_f32_16x16x32_bf16(false, a, false, bv, (short)0, cv, false, false);
  }
#pragma unroll
  for (int r = 0; r < 8; ++r) {
    const int m    = r + 8 * g;
    const int grow = row0 + m;           // global row in [0, B*N)
    const int col  = col0 + ln;
    qb[(size_t)grow * kD + col] = (bf16_t)cq[r];
    kb[(size_t)grow * kD + col] = (bf16_t)ck[r];
    const int bbi = grow >> 11;          // / kN
    const int nni = grow & (kN - 1);
    vt[((size_t)bbi * kD + col) * kN + nni] = (bf16_t)cv[r];  // V^T: [b][d][n]
  }
}

// ---------------- Prior association P ----------------
__global__ __launch_bounds__(256) void prior_kernel(const float* __restrict__ sig,
                                                    float* __restrict__ P) {
  const int row = blockIdx.x;          // b*N + i
  const int i   = row & (kN - 1);
  const float inv = 1.0f / sig[row];
  __shared__ float ev[kN];
  __shared__ float wred[8];
  __shared__ float stot;
  float part = 0.f;
  for (int j = threadIdx.x; j < kN; j += 256) {
    float d = (float)(i - j);
    float t = d * inv;
    float e = __expf(-0.5f * t * t);
    ev[j] = e;
    part += e;
  }
#pragma unroll
  for (int m = 16; m >= 1; m >>= 1) part += __shfl_xor(part, m, 32);
  if ((threadIdx.x & 31) == 0) wred[threadIdx.x >> 5] = part;
  __syncthreads();
  if (threadIdx.x == 0) {
    float t = 0.f;
#pragma unroll
    for (int w = 0; w < 8; ++w) t += wred[w];
    stot = t;
  }
  __syncthreads();
  const float rs = 1.0f / stot;
  float* prow = P + (size_t)row * kN;
  for (int j = threadIdx.x; j < kN; j += 256) prow[j] = ev[j] * rs;
}

// ---- async staging: copy one K tile (32x512) and one V tile (256x32) into LDS ----
// 128 threads cooperate; each b128 op moves 16 B/lane (512 B/wave). ASYNCcnt tracked.
__device__ __forceinline__ void stage_async(const bf16_t* __restrict__ Kb,
                                            const bf16_t* __restrict__ Vt,
                                            bf16_t* kdst, bf16_t* vdst,
                                            int b, int j0, int half, int t) {
  // K tile: contiguous 32 KB (rows j0..j0+31 of Kb[b])
  unsigned long long gk =
      (unsigned long long)(uintptr_t)(Kb + (size_t)(b * kN + j0) * kD) + (unsigned)(t * 16);
  unsigned int lk = (unsigned int)(uintptr_t)kdst + (unsigned)(t * 16);
#pragma unroll
  for (int i = 0; i < 16; ++i) {
    asm volatile("global_load_async_to_lds_b128 %0, %1, off"
                 :: "v"(lk), "v"(gk) : "memory");
    gk += 2048; lk += 2048;
  }
  // V tile: 256 rows (d = half*256 + r) x 32 keys, global row stride kN elements
  const int r0 = t >> 2;
  const int bo = (t & 3) * 16;
  unsigned long long gv =
      (unsigned long long)(uintptr_t)(Vt + ((size_t)(b * kD + half * 256 + r0)) * kN + j0) +
      (unsigned)bo;
  unsigned int lv = (unsigned int)(uintptr_t)vdst + (unsigned)(r0 * 64 + bo);
#pragma unroll
  for (int i = 0; i < 8; ++i) {
    asm volatile("global_load_async_to_lds_b128 %0, %1, off"
                 :: "v"(lv), "v"(gv) : "memory");
    gv += (unsigned long long)32 * kN * 2;   // advance 32 rows
    lv += 32 * 64;
  }
}

// ---------------- Flash attention: S = softmax(QK^T/sqrt(d)), Z = S V ----------------
// 4 waves/block share async-staged K/V tiles in LDS; Q fragments stay in VGPRs.
__global__ __launch_bounds__(128) void attn_kernel(
    const unsigned short* __restrict__ qb_, const unsigned short* __restrict__ kb_,
    const unsigned short* __restrict__ vt_, float* __restrict__ Z) {
  const bf16_t* Qb = (const bf16_t*)qb_;
  const bf16_t* Kb = (const bf16_t*)kb_;
  const bf16_t* Vt = (const bf16_t*)vt_;
  const int t    = threadIdx.x;
  const int wid  = t >> 5;
  const int lane = t & 31;
  const int wave = blockIdx.x * 4 + wid;            // 0..511
  const int half = blockIdx.y;                      // col half: [0,256) or [256,512)
  const int b  = wave / (kN / 16);
  const int q0 = (wave % (kN / 16)) * 16;
  const int g = lane >> 4, ln = lane & 15;

  extern __shared__ unsigned short smem_raw[];
  bf16_t* kbuf = (bf16_t*)smem_raw;                 // [2][32][512]
  bf16_t* vbuf = kbuf + 2 * KT_ELEM;                // [2][256][32]
  bf16_t* sPd  = vbuf + 2 * VT_ELEM;                // [4][16][32]

  // resident Q fragments: 16 chunks of 16x32 bf16
  const bf16_t* Qrow = Qb + (size_t)(b * kN + q0 + ln) * kD;
  v16bf aq[16];
#pragma unroll
  for (int kk = 0; kk < 16; ++kk) {
    v8bf alo = *(const v8bf*)(Qrow + kk * 32 + 16 * g);
    v8bf ahi = *(const v8bf*)(Qrow + kk * 32 + 16 * g + 8);
    aq[kk] = cat8(alo, ahi);
  }

  v8f zacc[16];
#pragma unroll
  for (int ct = 0; ct < 16; ++ct) zacc[ct] = {};
  float m_r[8], l_r[8];
#pragma unroll
  for (int r = 0; r < 8; ++r) { m_r[r] = -__builtin_inff(); l_r[r] = 0.f; }

  const float scale = 0.044194173824159216f;        // 1/sqrt(512)
  const int nblk = kN / 32;                         // 64

  // prologue: stage buffer 0
  stage_async(Kb, Vt, kbuf, vbuf, b, 0, half, t);
  asm volatile("s_wait_asynccnt 0" ::: "memory");
  __syncthreads();

  for (int jb = 0; jb < nblk; ++jb) {
    const int j0  = jb * 32;
    const int cur = jb & 1;
    const bf16_t* kc = kbuf + cur * KT_ELEM;
    const bf16_t* vc = vbuf + cur * VT_ELEM;

    // kick off next tile into the other buffer (overlaps with compute below)
    if (jb + 1 < nblk)
      stage_async(Kb, Vt, kbuf + (cur ^ 1) * KT_ELEM, vbuf + (cur ^ 1) * VT_ELEM,
                  b, j0 + 32, half, t);

    // ---- scores for keys [j0, j0+32): two 16x16 f32 tiles, K/V from LDS ----
    v8f c0 = {}, c1 = {};
#pragma unroll
    for (int kk = 0; kk < 16; ++kk) {
      v16bf b0 = *(const v16bf*)(kc + (size_t)ln * kD + kk * 32 + 16 * g);
      v16bf b1 = *(const v16bf*)(kc + (size_t)(16 + ln) * kD + kk * 32 + 16 * g);
      c0 = __builtin_amdgcn_wmma_f32_16x16x32_bf16(false, aq[kk], false, b0, (short)0, c0, false, false);
      c1 = __builtin_amdgcn_wmma_f32_16x16x32_bf16(false, aq[kk], false, b1, (short)0, c1, false, false);
    }

    // ---- online softmax update (row = reg r + 8g, spread over 16 lanes) ----
    float p0[8], p1[8];
#pragma unroll
    for (int r = 0; r < 8; ++r) {
      float s0 = c0[r] * scale, s1 = c1[r] * scale;
      float mx = fmaxf(s0, s1);
#pragma unroll
      for (int msk = 8; msk >= 1; msk >>= 1) mx = fmaxf(mx, __shfl_xor(mx, msk, 32));
      float mnew = fmaxf(m_r[r], mx);
      float e0 = __expf(s0 - mnew), e1 = __expf(s1 - mnew);
      float sum = e0 + e1;
#pragma unroll
      for (int msk = 8; msk >= 1; msk >>= 1) sum += __shfl_xor(sum, msk, 32);
      float corr = __expf(m_r[r] - mnew);
      l_r[r] = l_r[r] * corr + sum;
      m_r[r] = mnew;
      p0[r] = e0; p1[r] = e1;
#pragma unroll
      for (int ct = 0; ct < 16; ++ct) zacc[ct][r] = zacc[ct][r] * corr;
    }

    // ---- remap S tile (C-layout) -> A-fragment layout via LDS ----
    bf16_t* sw = sPd + wid * (16 * 32);
#pragma unroll
    for (int r = 0; r < 8; ++r) {
      sw[(r + 8 * g) * 32 + ln]      = (bf16_t)p0[r];
      sw[(r + 8 * g) * 32 + 16 + ln] = (bf16_t)p1[r];
    }
    asm volatile("s_wait_dscnt 0" ::: "memory");    // LDS RAW within wave
    v8bf plo = *(const v8bf*)(sw + ln * 32 + 16 * g);
    v8bf phi = *(const v8bf*)(sw + ln * 32 + 16 * g + 8);
    v16bf ap = cat8(plo, phi);

    // ---- Z += S * V over 16 col tiles (V tile rows contiguous in LDS) ----
#pragma unroll
    for (int ct = 0; ct < 16; ++ct) {
      v16bf bv = *(const v16bf*)(vc + (ct * 16 + ln) * 32 + 16 * g);
      zacc[ct] = __builtin_amdgcn_wmma_f32_16x16x32_bf16(false, ap, false, bv, (short)0,
                                                         zacc[ct], false, false);
    }

    // make sure this wave's async issues (for jb+1) completed, then publish buffers
    asm volatile("s_wait_asynccnt 0" ::: "memory");
    __syncthreads();
  }

  // ---- normalize and store ----
#pragma unroll
  for (int ct = 0; ct < 16; ++ct) {
    const int col = half * 256 + ct * 16 + ln;
#pragma unroll
    for (int r = 0; r < 8; ++r) {
      const int m = r + 8 * g;
      Z[(size_t)(b * kN + q0 + m) * kD + col] = zacc[ct][r] / l_r[r];
    }
  }
}

// ---------------- host launcher ----------------
extern "C" void kernel_launch(void* const* d_in, const int* in_sizes, int n_in,
                              void* d_out, int out_size, void* d_ws, size_t ws_size,
                              hipStream_t stream) {
  const float* x  = (const float*)d_in[0];
  const float* Wq = (const float*)d_in[1];
  const float* Wk = (const float*)d_in[2];
  const float* Wv = (const float*)d_in[3];
  const float* Ws = (const float*)d_in[4];
  float* Z = (float*)d_out;            // (B,N,D)
  float* P = Z + XN;                   // (B,N,N) follows Z

  // workspace layout (bf16 stored as ushort); ~35 MB total
  unsigned short* xb  = (unsigned short*)d_ws;
  unsigned short* wqt = xb  + XN;
  unsigned short* wkt = wqt + WN;
  unsigned short* wvt = wkt + WN;
  unsigned short* qb  = wvt + WN;
  unsigned short* kb  = qb  + XN;
  unsigned short* vt  = kb  + XN;
  float*          sig = (float*)(vt + XN);

  cvt_x_kernel<<<(unsigned)(XN / 256), 256, 0, stream>>>(x, xb);
  cvt_w_kernel<<<(unsigned)(WN / 256), 256, 0, stream>>>(Wq, Wk, Wv, wqt, wkt, wvt);
  sigma_kernel<<<(kB * kN) / 8, 256, 0, stream>>>(x, Ws, sig);
  proj_kernel<<<(kB * kN / 16) * (kD / 16) / 4, 128, 0, stream>>>(xb, wqt, wkt, wvt, qb, kb, vt);
  prior_kernel<<<kB * kN, 256, 0, stream>>>(sig, P);
  dim3 agrid((kB * kN / 16) / 4, 2);
  attn_kernel<<<agrid, 128, (unsigned)SMEM_BYTES, stream>>>(qb, kb, vt, Z);
}
